// ModulatedDeformConvPack_58617713656011
// MI455X (gfx1250) — matrix operand order repaired and checked by reading, not compiled, not run on hardware
//
#include <hip/hip_runtime.h>
#include <hip/hip_bf16.h>
#include <stdint.h>

// Modulated deformable conv (DCNv2) for gfx1250 (MI455X).
// B=2, C=64, H=W=128, O=64, KH=KW=3, stride=1, pad=1, dil=1, DG=1.
//
// Stage 1: fp32 offset conv (27 ch, coordinate-critical -> full precision).
// Stage 2: bilinear sample + sigmoid-mask, emit f16 panel S[n][kd] (n-major so
//          the WMMA B-fragment K-pairs are contiguous dwords).
// Stage 3: GEMM out[o][n] = sum_kd Wh[o][kd] * S[n][kd] via
//          v_wmma_f32_16x16x32_f16 (f16 in, f32 accumulate), bias in epilogue.

#define HI 128
#define WIMG 128
#define CI 64
#define OC 64
#define NPIX (HI * WIMG)          // 16384
#define NB 2
#define NTOT (NB * NPIX)          // 32768
#define KD (CI * 9)               // 576
#define KT (KD / 32)              // 18 WMMA K-steps

typedef __attribute__((ext_vector_type(16))) _Float16 v16h;
typedef __attribute__((ext_vector_type(8)))  float    v8f;

// ---------------------------------------------------------------- stage 0
__global__ void mdcn_cast_weight(const float* __restrict__ w,
                                 _Float16* __restrict__ wh, int n) {
  int i = blockIdx.x * 256 + threadIdx.x;
  if (i < n) wh[i] = (_Float16)w[i];   // weight.reshape(O,C,9) is contiguous
}

// ---------------------------------------------------------------- stage 1
// off[b][j][p] = conv3x3(x, offset_w)[b,j,p] + offset_b[j]   (fp32)
__global__ void mdcn_offset_conv(const float* __restrict__ x,
                                 const float* __restrict__ ow,
                                 const float* __restrict__ ob,
                                 float* __restrict__ off) {
  int idx = blockIdx.x * 256 + threadIdx.x;      // 2*27*16384 = 884736 exact
  int p = idx & (NPIX - 1);
  int t = idx >> 14;
  int j = t % 27;
  int b = t / 27;
  int h = p >> 7, w = p & 127;

  float acc = ob[j];
  const float* xb = x + (size_t)b * CI * NPIX;
  const float* wj = ow + (size_t)j * CI * 9;
#pragma unroll 4
  for (int c = 0; c < CI; ++c) {
    const float* xc = xb + c * NPIX;
    const float* wc = wj + c * 9;
#pragma unroll
    for (int ki = 0; ki < 3; ++ki) {
      int yy = h + ki - 1;
      if ((unsigned)yy < (unsigned)HI) {
#pragma unroll
        for (int kj = 0; kj < 3; ++kj) {
          int xx = w + kj - 1;
          if ((unsigned)xx < (unsigned)WIMG)
            acc = fmaf(xc[yy * WIMG + xx], wc[ki * 3 + kj], acc);
        }
      }
    }
  }
  off[idx] = acc;
}

// ---------------------------------------------------------------- stage 2
__device__ __forceinline__ float mdcn_corner(const float* __restrict__ xc,
                                             int yi, int xi) {
  bool v = ((unsigned)yi < (unsigned)HI) && ((unsigned)xi < (unsigned)WIMG);
  int y = yi < 0 ? 0 : (yi > HI - 1 ? HI - 1 : yi);
  int xq = xi < 0 ? 0 : (xi > WIMG - 1 ? WIMG - 1 : xi);
  float val = xc[y * WIMG + xq];
  return v ? val : 0.0f;
}

// One block = 4 pixels; tid/64 -> local pixel, tid%64 -> channel c.
// Writes S[n*576 + c*9 + k] as f16 (kd = c*9+k matches weight.reshape(O,C,9)).
__global__ void mdcn_sample(const float* __restrict__ x,
                            const float* __restrict__ off,
                            _Float16* __restrict__ S) {
  int tid = threadIdx.x;
  int lp = tid >> 6;               // local pixel 0..3
  int c = tid & 63;                // channel
  int n = blockIdx.x * 4 + lp;     // global pixel 0..32767
  int b = n >> 14;
  int p = n & (NPIX - 1);
  int h = p >> 7, w = p & 127;

  __shared__ float spy[4][9], spx[4][9], sm[4][9];
  if (c < 27) {
    int k = c % 9, grp = c / 9;    // 0-8: dy, 9-17: dx, 18-26: mask
    float v = off[((size_t)b * 27 + c) * NPIX + p];
    if (grp == 0)      spy[lp][k] = (float)(h - 1 + k / 3) + v;
    else if (grp == 1) spx[lp][k] = (float)(w - 1 + k % 3) + v;
    else               sm[lp][k]  = 1.0f / (1.0f + __expf(-v));
  }
  __syncthreads();

  const float* xc = x + ((size_t)b * CI + c) * NPIX;
  _Float16* Sp = S + (size_t)n * KD + c * 9;
#pragma unroll
  for (int k = 0; k < 9; ++k) {
    float py = spy[lp][k], px = spx[lp][k];
    float y0f = floorf(py), x0f = floorf(px);
    int y0 = (int)y0f, x0 = (int)x0f;
    float wy = py - y0f, wx = px - x0f;
    float v00 = mdcn_corner(xc, y0, x0);
    float v01 = mdcn_corner(xc, y0, x0 + 1);
    float v10 = mdcn_corner(xc, y0 + 1, x0);
    float v11 = mdcn_corner(xc, y0 + 1, x0 + 1);
    float s = (v00 * (1.0f - wy) * (1.0f - wx) + v01 * (1.0f - wy) * wx +
               v10 * wy * (1.0f - wx) + v11 * wy * wx) * sm[lp][k];
    Sp[k] = (_Float16)s;
  }
}

// ---------------------------------------------------------------- stage 3
// out[o][n] = sum_kd Wh[o][kd] * S[n][kd] + bias[o], via WMMA f16 -> f32.
// Block: 256 threads = 8 waves; wave (mt,nt): mt=wave&3 -> M tile (O rows),
// nt=wave>>2 -> N sub-tile. Block covers 64x32 outputs; grid = 32768/32.
__global__ void mdcn_gemm_wmma(const _Float16* __restrict__ Wh,
                               const _Float16* __restrict__ S,
                               const float* __restrict__ bias,
                               float* __restrict__ out) {
  int tid = threadIdx.x;
  int lane = tid & 31;
  int wv = tid >> 5;
  int mt = wv & 3;
  int nt = wv >> 2;
  int M0 = mt * 16;
  int hi = lane >> 4;              // lane half (0/1)
  int N = blockIdx.x * 32 + nt * 16 + (lane & 15);

  union Frag { v16h h; uint32_t u[8]; };
  v8f acc = {};

  int arow = M0 + (lane & 15);
  const uint32_t* Wrow = (const uint32_t*)(Wh + (size_t)arow * KD);
  const uint32_t* Scol = (const uint32_t*)(S + (size_t)N * KD);
  const int aBase = hi ? 8 : 0;    // A: lanes 16-31 hold K+8 (16-bit A layout)
  const int bBase = hi ? 16 : 0;   // unused: B reads with aBase, lanes 16-31 hold K+8

  for (int kt = 0; kt < KT; ++kt) {
    int koff = kt * 32;
    Frag a, bf;
#pragma unroll
    for (int vi = 0; vi < 8; ++vi) {
      // A fragment: K = koff + aBase + (vi>=4 ? 16 : 0) + 2*(vi&3)
      int ka = koff + aBase + ((vi & 4) ? 16 : 0) + ((vi & 3) * 2);
      a.u[vi] = Wrow[ka >> 1];
      // B fragment: same K as A: koff + aBase + (vi>=4 ? 16 : 0) + 2*(vi&3)
      int kb = koff + aBase + ((vi & 4) ? 16 : 0) + ((vi & 3) * 2);
      bf.u[vi] = Scol[kb >> 1];
    }
    acc = __builtin_amdgcn_wmma_f32_16x16x32_f16(
        /*neg_a=*/false, a.h, /*neg_b=*/false, bf.h,
        /*c_mod=*/(short)0, acc, /*reuse_a=*/false, /*reuse_b=*/false);
  }

  // D layout: VGPR i -> M = M0 + i + 8*hi, N = column held by this lane.
  int b = N >> 14;
  int p = N & (NPIX - 1);
#pragma unroll
  for (int i = 0; i < 8; ++i) {
    int o = M0 + i + (hi ? 8 : 0);
    out[((size_t)(b * OC + o)) * NPIX + p] = acc[i] + bias[o];
  }
}

// ---------------------------------------------------------------- launch
extern "C" void kernel_launch(void* const* d_in, const int* in_sizes, int n_in,
                              void* d_out, int out_size, void* d_ws,
                              size_t ws_size, hipStream_t stream) {
  const float* x        = (const float*)d_in[0];   // (2,64,128,128)
  const float* weight   = (const float*)d_in[1];   // (64,64,3,3)
  const float* bias     = (const float*)d_in[2];   // (64,)
  const float* offset_w = (const float*)d_in[3];   // (27,64,3,3)
  const float* offset_b = (const float*)d_in[4];   // (27,)
  float* out = (float*)d_out;                      // (2,64,128,128)

  // Workspace layout (all offsets 256B-aligned):
  //   off : fp32  [2][27][16384]          = 3,538,944 B
  //   Wh  : f16   [64][576]               =    73,728 B
  //   S   : f16   [32768][576] (n-major)  = 37,748,736 B
  char* ws = (char*)d_ws;
  float*     off = (float*)ws;
  _Float16*  Wh  = (_Float16*)(ws + 3538944);
  _Float16*  S   = (_Float16*)(ws + 3538944 + 73728);

  mdcn_cast_weight<<<(OC * KD + 255) / 256, 256, 0, stream>>>(weight, Wh,
                                                              OC * KD);
  mdcn_offset_conv<<<(NB * 27 * NPIX) / 256, 256, 0, stream>>>(x, offset_w,
                                                               offset_b, off);
  mdcn_sample<<<NTOT / 4, 256, 0, stream>>>(x, off, S);
  mdcn_gemm_wmma<<<NTOT / 32, 256, 0, stream>>>(Wh, S, bias, out);
}
